// GraphConvolution_79474074845612
// MI455X (gfx1250) — matrix operand order repaired and checked by reading, not compiled, not run on hardware
//
#include <hip/hip_runtime.h>
#include <hip/hip_bf16.h>
#include <hip/hip_fp16.h>

// ---------------- problem constants ----------------
#define NS    3000          // stocks
#define NPAD  3008          // padded to multiple of 16 (188 tiles)
#define HD    128           // hidden
#define PP    16            // d_piece
#define EE    4             // heads
#define RRR   3             // relations
#define KP    (PP*HD)       // 2048  (tt inner feature dim)

// LDS pitches (in 16-bit elements); 40*2B = 80B rows keep b128 alignment.
// 40 = 32 data halfs + 8 pad halfs == TDM pad_interval 16 DW / pad_amount 4 DW.
#define APITCH 40
#define BPITCH 40

typedef __attribute__((ext_vector_type(16))) _Float16 v16h;
typedef __attribute__((ext_vector_type(16))) __bf16   v16bf;
typedef __attribute__((ext_vector_type(8)))  float    v8f;
typedef unsigned int v4u  __attribute__((ext_vector_type(4)));
typedef int          v8i_t __attribute__((ext_vector_type(8)));
typedef int          v4i_t __attribute__((ext_vector_type(4)));

union FragBF { uint4 u[2]; v16bf v; };
union FragH  { uint4 u[2]; v16h  v; };
union Pack8BF { uint4 u; __bf16  h[8]; };

// ---------------- Tensor Data Mover support ----------------
#if defined(__has_builtin)
# if __has_builtin(__builtin_amdgcn_tensor_load_to_lds)
#  define HAVE_TDM 1
# endif
#endif
#ifndef HAVE_TDM
# define HAVE_TDM 0
#endif

// DMA a 2D tile of 16-bit data: tile = 32 elems wide x 128 rows, row stride
// `stride_elems` in global, written to LDS with +8-half padding per row
// (pad_interval=16 DWORDs -> code 3, pad_amount=4 DWORDs -> code 3), giving
// an LDS row pitch of 40 halfs == APITCH/BPITCH.  tensor_dim0/1 bound the
// valid region measured from the tile origin; OOB reads return zero (free
// edge guards).  Issued once per wave (EXEC ignored); caller gates on wave 0
// and waits TENSORcnt before the workgroup barrier.
__device__ __forceinline__ void tdm_load_tile16(
    unsigned lds_off, const void* gp,
    unsigned tens_w, unsigned tens_h, unsigned stride_elems)
{
#if HAVE_TDM
  unsigned long long ga = (unsigned long long)(uintptr_t)gp;
  v4u g0;
  g0.x = 1u;                                                // count=1, user D#
  g0.y = lds_off;                                           // lds_addr (bytes)
  g0.z = (unsigned)(ga & 0xFFFFFFFFu);                      // global_addr[31:0]
  g0.w = (unsigned)((ga >> 32) & 0x1FFFFFFu) | (2u << 30);  // [56:32] | type=2
  v8i_t g1;
  g1[0] = (int)((1u << 16) | (1u << 20) | (3u << 22) | (3u << 25));
          // data_size=2B | pad_enable | pad_interval=16DW | pad_amount=4DW
  g1[1] = (int)((tens_w & 0xFFFFu) << 16);                  // tensor_dim0 lo
  g1[2] = (int)(((tens_w >> 16) & 0xFFFFu) | ((tens_h & 0xFFFFu) << 16));
  g1[3] = (int)(((tens_h >> 16) & 0xFFFFu) | (32u << 16));  // tile_dim0 = 32
  g1[4] = (int)128u;                                        // tile_dim1 = 128
  g1[5] = (int)stride_elems;                                // dim0 stride lo
  g1[6] = 0;
  g1[7] = 0;                                                // dim1 stride (2D)
  v4i_t gz = {0, 0, 0, 0};
# if defined(__clang_major__) && __clang_major__ >= 23
  v8i_t gz8 = {0, 0, 0, 0, 0, 0, 0, 0};
  __builtin_amdgcn_tensor_load_to_lds(g0, g1, gz, gz, gz8, 0);
# else
  __builtin_amdgcn_tensor_load_to_lds(g0, g1, gz, gz, 0);
# endif
#else
  (void)lds_off; (void)gp; (void)tens_w; (void)tens_h; (void)stride_elems;
#endif
}
__device__ __forceinline__ void tdm_wait() {
#if HAVE_TDM
  __builtin_amdgcn_s_wait_tensorcnt(0);
#endif
}

// =====================================================================
// Generic bf16 WMMA GEMM:  C[M,Nc] = A[M,K] @ B[K,Nc]
//   AMODE 0: A is bf16 in global (lda elems/row); A tile staged by TDM.
//   AMODE 1: A tile = elementwise product of `adjcnt` binary fp32 [NS,NS]
//            adjacency matrices starting at adjbase (fused, never stored).
// Block: 256 thr = 8 waves; tile 128 rows x 64 cols; wave = 16 rows, 4 frags.
// A-fragment per ISA layout: lane(m16,hi) reads K[hi*8..+8) and K[16+hi*8..+8)
// => two contiguous b128 from a natural-order LDS row. Same for B.
// =====================================================================
template<int AMODE>
__global__ __launch_bounds__(256) void gemm_wmma_bf16(
    const __bf16* __restrict__ A, int lda, int apad,
    const float*  __restrict__ adjbase, int adjcnt,
    const __bf16* __restrict__ B, int ldb, int K,
    float* __restrict__ Cf, __bf16* __restrict__ Cb,
    _Float16* __restrict__ ChT, int ldT,
    int ldc, int mstore,
    const float* __restrict__ biasVec,
    const float* __restrict__ biasFull, int biasRows,
    int relu)
{
  __shared__ __bf16 As[128 * APITCH];
  __shared__ __bf16 Bs[64 * BPITCH];
  const int t    = threadIdx.x;
  const int wv   = t >> 5, lane = t & 31;           // wave32
  const int m16  = lane & 15, hi = lane >> 4;
  const int row0 = blockIdx.x * 128;
  const int n0   = blockIdx.y * 64;
  const bool use_tdm = (HAVE_TDM != 0) && (AMODE == 0);

  v8f zero8 = {0.f,0.f,0.f,0.f,0.f,0.f,0.f,0.f};
  v8f acc[4]; acc[0]=zero8; acc[1]=zero8; acc[2]=zero8; acc[3]=zero8;

  for (int k0 = 0; k0 < K; k0 += 32) {
    // ---- stage A tile (128 x 32), natural K order ----
    if (use_tdm) {
      if (wv == 0) {
        unsigned th = (apad > row0) ? (unsigned)(apad - row0) : 0u;
        tdm_load_tile16((unsigned)(uintptr_t)&As[0],
                        A + (size_t)row0 * lda + k0,
                        (unsigned)(K - k0), th, (unsigned)lda);
      }
    } else {
      #pragma unroll
      for (int cidx = t; cidx < 512; cidx += 256) {
        int r = cidx >> 2, cc = cidx & 3;
        int gr = row0 + r;
        Pack8BF pk;
        if (AMODE == 0) {
          uint4 u = {0u,0u,0u,0u};
          if (gr < apad) u = *(const uint4*)(A + (size_t)gr * lda + k0 + cc * 8);
          pk.u = u;
        } else {
          float f[8];
          #pragma unroll
          for (int q = 0; q < 8; ++q) f[q] = 0.f;
          int gk = k0 + cc * 8;
          if (gr < NS && gk + 8 <= NS) {
            const float* p0 = adjbase + (size_t)gr * NS + gk;
            float4 x0 = *(const float4*)p0;
            float4 x1 = *(const float4*)(p0 + 4);
            f[0]=x0.x; f[1]=x0.y; f[2]=x0.z; f[3]=x0.w;
            f[4]=x1.x; f[5]=x1.y; f[6]=x1.z; f[7]=x1.w;
            for (int m = 1; m < adjcnt; ++m) {        // fused ∏ adj (binary)
              const float* pm = p0 + (size_t)m * NS * NS;
              float4 y0 = *(const float4*)pm;
              float4 y1 = *(const float4*)(pm + 4);
              f[0]*=y0.x; f[1]*=y0.y; f[2]*=y0.z; f[3]*=y0.w;
              f[4]*=y1.x; f[5]*=y1.y; f[6]*=y1.z; f[7]*=y1.w;
            }
            __builtin_prefetch(p0 + 32, 0, 1);
          }
          #pragma unroll
          for (int q = 0; q < 8; ++q) pk.h[q] = (__bf16)f[q];
        }
        *(uint4*)(As + r * APITCH + cc * 8) = pk.u;
      }
    }
    // ---- stage B tile (32 x 64), transposed: Bs[n][k] (overlaps TDM) ----
    {
      int kk = t >> 3, cc = t & 7;
      uint4 u = *(const uint4*)(B + (size_t)(k0 + kk) * ldb + n0 + cc * 8);
      Pack8BF pk; pk.u = u;
      #pragma unroll
      for (int q = 0; q < 8; ++q) Bs[(cc * 8 + q) * BPITCH + kk] = pk.h[q];
    }
    if (use_tdm && wv == 0) tdm_wait();
    __syncthreads();
    FragBF fa;
    fa.u[0] = *(const uint4*)(As + (wv * 16 + m16) * APITCH + hi * 8);
    fa.u[1] = *(const uint4*)(As + (wv * 16 + m16) * APITCH + 16 + hi * 8);
    #pragma unroll
    for (int nt = 0; nt < 4; ++nt) {
      FragBF fb;
      fb.u[0] = *(const uint4*)(Bs + (nt * 16 + m16) * BPITCH + hi * 8);
      fb.u[1] = *(const uint4*)(Bs + (nt * 16 + m16) * BPITCH + 16 + hi * 8);
      acc[nt] = __builtin_amdgcn_wmma_f32_16x16x32_bf16(
          false, fa.v, false, fb.v, (short)0, acc[nt], false, false);
    }
    __syncthreads();
  }
  // ---- epilogue: C frag layout M = q + 8*hi, N = m16 ----
  #pragma unroll
  for (int nt = 0; nt < 4; ++nt) {
    #pragma unroll
    for (int q = 0; q < 8; ++q) {
      int gr = row0 + wv * 16 + hi * 8 + q;
      int gc = n0 + nt * 16 + m16;
      float val = acc[nt][q];
      if (biasVec)  val += biasVec[gc];
      if (biasFull && gr < biasRows) val += biasFull[(size_t)gr * ldc + gc];
      if (relu) val = fmaxf(val, 0.f);
      if (gr < mstore) {
        if (Cf)  Cf[(size_t)gr * ldc + gc] = val;
        if (Cb)  Cb[(size_t)gr * ldc + gc] = (__bf16)val;
        if (ChT) ChT[(size_t)gc * ldT + gr] = (_Float16)val;   // transposed f16
      }
    }
  }
}

// =====================================================================
// Fused masked-softmax + (attn @ v) with fp16 WMMA.
// grid (NPAD/16 row-tiles, E). Block 256 = 8 waves; wave w owns h-tile w.
// Accumulates over all 3 relations into one C fragment (no atomics).
// f1 cancels exactly in softmax (additive per row), so only f2 is used.
// V blocks (128h x 32j, stride NPAD) are DMA'd by the TDM while the VALU
// builds the prob tile.
// =====================================================================
__global__ __launch_bounds__(256) void attn_kernel(
    const float* __restrict__ adj,       // [R][NS][NS] binary
    const float* __restrict__ f2,        // [R][E][NPAD]
    const _Float16* __restrict__ vT,     // [R][E][HD][NPAD]  (h-major)
    __bf16* __restrict__ aggCb,          // [NPAD][E*HD]  Σ_r agg, cat layout
    float invscale)
{
  __shared__ _Float16 Ps[16 * APITCH];   // attn probs tile (natural order)
  __shared__ _Float16 Vs[128 * BPITCH];  // v block, [h][k]
  __shared__ float red_m[16][17], red_s[16][17];
  __shared__ float rowmax[16], rowinv[16];

  const int t = threadIdx.x, wv = t >> 5, lane = t & 31;
  const int m16 = lane & 15, hi = lane >> 4;
  const int i0 = blockIdx.x * 16;
  const int e  = blockIdx.y;

  v8f c = {0.f,0.f,0.f,0.f,0.f,0.f,0.f,0.f};

  for (int r = 0; r < RRR; ++r) {
    const float* f2p  = f2 + ((size_t)(r * EE) + e) * NPAD;
    const float* adjr = adj + (size_t)r * NS * NS;
    // ---------- pass 1: online row max & exp-sum over valid j ----------
    {
      int ir = t >> 4, lt = t & 15;
      int gi = i0 + ir;
      float m = -3.0e38f, s = 0.f;
      if (gi < NS) {
        const float* arow = adjr + (size_t)gi * NS;
        for (int j = lt; j < NS; j += 16) {
          float a = arow[j];
          if (a != 0.f || j == gi) {        // adj_ = clamp(adj + I) ∈ {0,1}
            float g = f2p[j] * invscale;
            if (g > m) { s = s * __expf(m - g) + 1.f; m = g; }
            else       { s += __expf(g - m); }
          }
          __builtin_prefetch(arow + j + 256, 0, 1);
        }
      }
      red_m[ir][lt] = m; red_s[ir][lt] = s;
      __syncthreads();
      if (t < 16) {
        float mm = -3.0e38f;
        for (int q = 0; q < 16; ++q) mm = fmaxf(mm, red_m[t][q]);
        float ss = 0.f;
        for (int q = 0; q < 16; ++q) ss += red_s[t][q] * __expf(red_m[t][q] - mm);
        rowmax[t] = mm;
        rowinv[t] = (ss > 0.f) ? 1.f / ss : 0.f;
      }
      __syncthreads();
    }
    // ---------- pass 2: build prob tiles, WMMA against v ----------
    const _Float16* vbase = vT + ((size_t)(r * EE) + e) * HD * NPAD;
    for (int jb = 0; jb < NPAD; jb += 32) {
#if HAVE_TDM
      if (wv == 0)
        tdm_load_tile16((unsigned)(uintptr_t)&Vs[0], vbase + jb,
                        (unsigned)(NPAD - jb), (unsigned)HD, (unsigned)NPAD);
#else
      #pragma unroll
      for (int idx = t; idx < 512; idx += 256) {    // v block 32j x 128h
        int h = idx >> 2, cc = idx & 3;
        uint4 u = *(const uint4*)(vbase + (size_t)h * NPAD + jb + cc * 8);
        *(uint4*)(Vs + h * BPITCH + cc * 8) = u;
      }
#endif
      // prob tile: branchless exp (co-executes with the TDM DMA)
      #pragma unroll
      for (int idx = t; idx < 512; idx += 256) {    // probs 16x32 -> fp16
        int ii = idx >> 5, jj = idx & 31;
        int gi = i0 + ii, gj = jb + jj;
        float a = 0.f;
        if (gi < NS && gj < NS) a = adjr[(size_t)gi * NS + gj];
        float p = __expf(f2p[gj] * invscale - rowmax[ii]) * rowinv[ii];
        bool valid = (a != 0.f) || (gj == gi);      // rowinv==0 kills pad rows
        Ps[ii * APITCH + jj] = (_Float16)(valid ? p : 0.f);
      }
#if HAVE_TDM
      if (wv == 0) tdm_wait();
#endif
      __syncthreads();
      FragH fa, fb;
      fa.u[0] = *(const uint4*)(Ps + m16 * APITCH + hi * 8);
      fa.u[1] = *(const uint4*)(Ps + m16 * APITCH + 16 + hi * 8);
      fb.u[0] = *(const uint4*)(Vs + (wv * 16 + m16) * BPITCH + hi * 8);
      fb.u[1] = *(const uint4*)(Vs + (wv * 16 + m16) * BPITCH + 16 + hi * 8);
      c = __builtin_amdgcn_wmma_f32_16x16x32_f16(
          false, fa.v, false, fb.v, (short)0, c, false, false);
      __syncthreads();
    }
  }
  #pragma unroll
  for (int q = 0; q < 8; ++q) {
    int gr = i0 + hi * 8 + q;
    int gc = e * HD + wv * 16 + m16;
    aggCb[(size_t)gr * (EE * HD) + gc] = (__bf16)c[q];   // cat layout [n][(e,h)]
  }
}

// ========================= small prep kernels =========================
__global__ void k_xb(const float* __restrict__ x, __bf16* __restrict__ xb) {
  int idx = blockIdx.x * blockDim.x + threadIdx.x;
  if (idx >= NPAD * HD) return;
  int n = idx >> 7;
  xb[idx] = (n < NS) ? (__bf16)x[(size_t)n * HD + (idx & 127)] : (__bf16)0.f;
}
__global__ void k_transpose_bf16(__bf16* __restrict__ dst, const float* __restrict__ src,
                                 int rows, int cols) {
  int idx = blockIdx.x * blockDim.x + threadIdx.x;
  if (idx >= rows * cols) return;
  int i = idx / cols, j = idx % cols;
  dst[(size_t)j * rows + i] = (__bf16)src[idx];
}
__global__ void k_convert_bf16(__bf16* __restrict__ dst, const float* __restrict__ src, int n) {
  int idx = blockIdx.x * blockDim.x + threadIdx.x;
  if (idx < n) dst[idx] = (__bf16)src[idx];
}
// relB[r][h][p*128+k] = rel1[p,h,k] * relu(map_r[p,h,k])
__global__ void k_relB(const float* __restrict__ rel1, const float* __restrict__ m1,
                       const float* __restrict__ m2, const float* __restrict__ m3,
                       __bf16* __restrict__ relB) {
  int idx = blockIdx.x * blockDim.x + threadIdx.x;
  if (idx >= RRR * PP * HD * HD) return;
  int k = idx & 127, h = (idx >> 7) & 127, p = (idx >> 14) & 15, r = idx >> 18;
  const float* mp = (r == 0) ? m1 : (r == 1) ? m2 : m3;
  size_t si = ((size_t)p * HD + h) * HD + k;
  float v = rel1[si] * fmaxf(mp[si], 0.f);
  relB[(size_t)r * HD * KP + (size_t)h * KP + p * HD + k] = (__bf16)v;
}
// wk_eff[e,h] = Σ_k Wk[e,k,h] * f2w[e,k]
__global__ void k_wk_eff(const float* __restrict__ Wk, const float* __restrict__ f2w,
                         float* __restrict__ wk_eff) {
  int idx = blockIdx.x * blockDim.x + threadIdx.x;
  if (idx >= EE * HD) return;
  int e = idx >> 7, h = idx & 127;
  float a = 0.f;
  for (int k = 0; k < HD; ++k) a += Wk[((size_t)e * HD + k) * HD + h] * f2w[e * HD + k];
  wk_eff[idx] = a;
}
// bk_eff[e,n] = Σ_k bk[e,n,k]*f2w[e,k] + f2b[e]
__global__ void k_bk_eff(const float* __restrict__ bk, const float* __restrict__ f2w,
                         const float* __restrict__ f2b, float* __restrict__ bk_eff) {
  int idx = blockIdx.x * blockDim.x + threadIdx.x;
  if (idx >= EE * NPAD) return;
  int e = idx / NPAD, n = idx % NPAD;
  float a = 0.f;
  if (n < NS) {
    const float* row = bk + ((size_t)e * NS + n) * HD;
    for (int k = 0; k < HD; ++k) a += row[k] * f2w[e * HD + k];
    a += f2b[e];
  }
  bk_eff[idx] = a;
}
// Weff[(e,h),o] = Σ_k ffnW[e,k,h] * Wsl[o, e*128+k]   (fold ffn+concat+Wsl)
__global__ void k_weff(const float* __restrict__ ffnW, const float* __restrict__ Wsl,
                       __bf16* __restrict__ WeffB) {
  int idx = blockIdx.x * blockDim.x + threadIdx.x;
  if (idx >= EE * HD * HD) return;
  int o = idx & 127, f = idx >> 7;             // f = e*128 + h
  int e = f >> 7, h = f & 127;
  float a = 0.f;
  for (int k = 0; k < HD; ++k)
    a += ffnW[((size_t)e * HD + k) * HD + h] * Wsl[(size_t)o * (EE * HD) + e * HD + k];
  WeffB[(size_t)f * HD + o] = (__bf16)a;
}
// beff[o] = bsl[o] + R * Σ_{e,k} ffnb[e,k] * Wsl[o, e*128+k]
__global__ void k_beff(const float* __restrict__ ffnb, const float* __restrict__ Wsl,
                       const float* __restrict__ bsl, float* __restrict__ beff) {
  int o = blockIdx.x * blockDim.x + threadIdx.x;
  if (o >= HD) return;
  float a = bsl[o];
  for (int f = 0; f < EE * HD; ++f) a += 3.f * ffnb[f] * Wsl[(size_t)o * (EE * HD) + f];
  beff[o] = a;
}
// f2[r,e,n] = tf[r,n,:]·wk_eff[e] + bk_eff[e,n]
__global__ void k_f2(const float* __restrict__ tf, const float* __restrict__ wk_eff,
                     const float* __restrict__ bk_eff, float* __restrict__ f2) {
  int idx = blockIdx.x * blockDim.x + threadIdx.x;
  if (idx >= RRR * EE * NPAD) return;
  int n = idx % NPAD, re = idx / NPAD, e = re % EE, r = re / EE;
  float a = 0.f;
  if (n < NS) {
    const float* trow = tf + ((size_t)r * NPAD + n) * HD;
    const float* w = wk_eff + e * HD;
    for (int h = 0; h < HD; ++h) a += trow[h] * w[h];
    a += bk_eff[e * NPAD + n];
  }
  f2[idx] = a;
}
// tf[r] = relu( (Σ_k s·tt)@W3 + [fl1|fl2] + b1 );   one block per row n
__global__ __launch_bounds__(128) void k_tf(
    const float* __restrict__ s, const __bf16* __restrict__ tt,
    const float* __restrict__ fl1, const float* __restrict__ fl2,
    const float* __restrict__ W3, const float* __restrict__ b1,
    float* __restrict__ tfr, __bf16* __restrict__ tfbr)
{
  int n = blockIdx.x, h = threadIdx.x;
  __shared__ float srow[128];
  __shared__ float part[16][9];
  __shared__ float ftp[16];
  srow[h] = s[(size_t)n * HD + h];
  __syncthreads();
  int p = h >> 3, kg = h & 7;
  float a = 0.f;
  const __bf16* trow = tt + (size_t)n * KP + p * HD;
  for (int k = kg; k < HD; k += 8) a += srow[k] * (float)trow[k];
  part[p][kg] = a;
  __syncthreads();
  if (h < 16) { float z = 0.f; for (int q = 0; q < 8; ++q) z += part[h][q]; ftp[h] = z; }
  __syncthreads();
  float val = 0.f;
  if (n < NS) {
    float ft = 0.f;
    for (int q = 0; q < PP; ++q) ft += ftp[q] * W3[q * HD + h];
    float fl = (h < 64) ? fl1[(size_t)n * 64 + h] : fl2[(size_t)n * 64 + (h - 64)];
    val = fmaxf(ft + fl + b1[(size_t)n * HD + h], 0.f);
  }
  tfr[(size_t)n * HD + h] = val;
  tfbr[(size_t)n * HD + h] = (__bf16)val;
}

// ========================= host-side launcher =========================
static inline int ceildiv(int a, int b) { return (a + b - 1) / b; }
static inline char* wsal(char*& p, size_t bytes) {
  char* r = p; p += (bytes + 255) & ~(size_t)255; return r;
}
static void gemm0(hipStream_t s, const __bf16* A, int lda, const __bf16* B, int ldb,
                  int K, int Nc, float* Cf, __bf16* Cb, _Float16* ChT, int ldT,
                  int ldc, int mstore, const float* bvec, const float* bfull,
                  int brows, int relu) {
  dim3 g(ceildiv(NPAD, 128), Nc / 64), b(256);
  gemm_wmma_bf16<0><<<g, b, 0, s>>>(A, lda, NPAD, nullptr, 0, B, ldb, K,
                                    Cf, Cb, ChT, ldT, ldc, mstore, bvec, bfull, brows, relu);
}
static void gemm_adj(hipStream_t s, const float* adjbase, int cnt, const __bf16* B,
                     int ldb, int Nc, float* Cf, int ldc) {
  dim3 g(ceildiv(NPAD, 128), Nc / 64), b(256);
  gemm_wmma_bf16<1><<<g, b, 0, s>>>(nullptr, 0, 0, adjbase, cnt, B, ldb, NPAD,
                                    Cf, nullptr, nullptr, 0, ldc, NPAD,
                                    nullptr, nullptr, 0, 0);
}

extern "C" void kernel_launch(void* const* d_in, const int* in_sizes, int n_in,
                              void* d_out, int out_size, void* d_ws, size_t ws_size,
                              hipStream_t stream) {
  const float* x    = (const float*)d_in[0];
  const float* adj  = (const float*)d_in[1];
  const float* Wm1  = (const float*)d_in[2];
  const float* Wm2  = (const float*)d_in[3];
  const float* rel1 = (const float*)d_in[4];
  const float* map1 = (const float*)d_in[5];
  const float* map2 = (const float*)d_in[6];
  const float* map3 = (const float*)d_in[7];
  const float* W1   = (const float*)d_in[8];
  const float* W2   = (const float*)d_in[9];
  const float* W3   = (const float*)d_in[10];
  const float* b1   = (const float*)d_in[11];
  // Wq/bq/f1w/f1b (12,13,18,19) are dead: f1 cancels in the row softmax.
  const float* Wk   = (const float*)d_in[14];
  const float* bk   = (const float*)d_in[15];
  const float* Wv   = (const float*)d_in[16];
  const float* bv   = (const float*)d_in[17];
  const float* f2w  = (const float*)d_in[20];
  const float* f2b  = (const float*)d_in[21];
  const float* ffnW = (const float*)d_in[22];
  const float* ffnb = (const float*)d_in[23];
  const float* Wsl  = (const float*)d_in[24];
  const float* bsl  = (const float*)d_in[25];
  float* out = (float*)d_out;

  char* p = (char*)d_ws;
  __bf16*   xb     = (__bf16*)wsal(p, (size_t)NPAD * HD * 2);
  __bf16*   Wm1T   = (__bf16*)wsal(p, (size_t)HD * HD * 2);
  __bf16*   Wm2T   = (__bf16*)wsal(p, (size_t)HD * HD * 2);
  __bf16*   W1b    = (__bf16*)wsal(p, (size_t)HD * 64 * 2);
  __bf16*   W2b    = (__bf16*)wsal(p, (size_t)HD * 64 * 2);
  __bf16*   WvTb   = (__bf16*)wsal(p, (size_t)EE * HD * HD * 2);
  __bf16*   relB   = (__bf16*)wsal(p, (size_t)RRR * HD * KP * 2);
  __bf16*   newsb  = (__bf16*)wsal(p, (size_t)NPAD * HD * 2);
  __bf16*   mktb   = (__bf16*)wsal(p, (size_t)NPAD * HD * 2);
  float*    fl1    = (float*)wsal(p, (size_t)NPAD * 64 * 4);
  __bf16*   fl2b   = (__bf16*)wsal(p, (size_t)NPAD * 64 * 2);
  __bf16*   tt     = (__bf16*)wsal(p, (size_t)NPAD * KP * 2);      // per-r reuse
  float*    sbuf   = (float*)wsal(p, (size_t)NPAD * HD * 4);
  float*    fl2    = (float*)wsal(p, (size_t)NPAD * 64 * 4);
  float*    tf     = (float*)wsal(p, (size_t)RRR * NPAD * HD * 4);
  __bf16*   tfb    = (__bf16*)wsal(p, (size_t)RRR * NPAD * HD * 2);
  float*    wk_eff = (float*)wsal(p, (size_t)EE * HD * 4);
  float*    bk_eff = (float*)wsal(p, (size_t)EE * NPAD * 4);
  float*    f2     = (float*)wsal(p, (size_t)RRR * EE * NPAD * 4);
  _Float16* vT     = (_Float16*)wsal(p, (size_t)RRR * EE * HD * NPAD * 2);
  __bf16*   aggCb  = (__bf16*)wsal(p, (size_t)NPAD * EE * HD * 2);
  __bf16*   WeffB  = (__bf16*)wsal(p, (size_t)EE * HD * HD * 2);
  float*    beff   = (float*)wsal(p, (size_t)HD * 4);

  dim3 b256(256);
  // ---- tiny prep ----
  k_xb<<<ceildiv(NPAD * HD, 256), b256, 0, stream>>>(x, xb);
  k_transpose_bf16<<<ceildiv(HD * HD, 256), b256, 0, stream>>>(Wm1T, Wm1, HD, HD);
  k_transpose_bf16<<<ceildiv(HD * HD, 256), b256, 0, stream>>>(Wm2T, Wm2, HD, HD);
  k_convert_bf16<<<ceildiv(HD * 64, 256), b256, 0, stream>>>(W1b, W1, HD * 64);
  k_convert_bf16<<<ceildiv(HD * 64, 256), b256, 0, stream>>>(W2b, W2, HD * 64);
  for (int e = 0; e < EE; ++e)
    k_transpose_bf16<<<ceildiv(HD * HD, 256), b256, 0, stream>>>(
        WvTb + (size_t)e * HD * HD, Wv + (size_t)e * HD * HD, HD, HD);
  k_relB<<<ceildiv(RRR * PP * HD * HD, 256), b256, 0, stream>>>(rel1, map1, map2, map3, relB);
  k_wk_eff<<<ceildiv(EE * HD, 256), b256, 0, stream>>>(Wk, f2w, wk_eff);
  k_bk_eff<<<ceildiv(EE * NPAD, 256), b256, 0, stream>>>(bk, f2w, f2b, bk_eff);
  k_weff<<<ceildiv(EE * HD * HD, 256), b256, 0, stream>>>(ffnW, Wsl, WeffB);
  k_beff<<<1, 128, 0, stream>>>(ffnb, Wsl, bsl, beff);

  // ---- Graph_Tensor ----
  gemm0(stream, xb, HD, Wm1T, HD, HD, HD, nullptr, newsb, nullptr, 0, HD, NPAD, nullptr, nullptr, 0, 0);
  gemm0(stream, xb, HD, Wm2T, HD, HD, HD, nullptr, mktb,  nullptr, 0, HD, NPAD, nullptr, nullptr, 0, 0);
  gemm0(stream, newsb, HD, W1b, 64, HD, 64, fl1, nullptr, nullptr, 0, 64, NPAD, nullptr, nullptr, 0, 0);
  gemm0(stream, mktb,  HD, W2b, 64, HD, 64, nullptr, fl2b, nullptr, 0, 64, NPAD, nullptr, nullptr, 0, 0);
  for (int r = 0; r < RRR; ++r) {
    gemm0(stream, newsb, HD, relB + (size_t)r * HD * KP, KP, HD, KP,
          nullptr, tt, nullptr, 0, KP, NPAD, nullptr, nullptr, 0, 0);     // tt_r
    gemm_adj(stream, adj, r + 1, mktb, HD, HD, sbuf, HD);                 // s = ∏adj @ mkt
    gemm_adj(stream, adj + (size_t)r * NS * NS, 1, fl2b, 64, 64, fl2, 64);// fl2 = adj_r @ fl2b
    k_tf<<<NPAD, 128, 0, stream>>>(sbuf, tt, fl1, fl2, W3, b1,
                                   tf + (size_t)r * NPAD * HD,
                                   tfb + (size_t)r * NPAD * HD);
  }

  // ---- attention inputs ----
  k_f2<<<ceildiv(RRR * EE * NPAD, 256), b256, 0, stream>>>(tf, wk_eff, bk_eff, f2);
  for (int r = 0; r < RRR; ++r)
    for (int e = 0; e < EE; ++e)
      gemm0(stream, tfb + (size_t)r * NPAD * HD, HD, WvTb + (size_t)e * HD * HD, HD,
            HD, HD, nullptr, nullptr,
            vT + (size_t)(r * EE + e) * HD * NPAD, NPAD,
            HD, NPAD, nullptr, bv + (size_t)e * NS * HD, NS, 0);

  // ---- fused softmax + attn@v (fp16 WMMA), Σ over relations ----
  {
    dim3 g(NPAD / 16, EE);
    attn_kernel<<<g, b256, 0, stream>>>(adj, f2, vT, aggCb, 1.0f / sqrtf((float)HD));
  }

  // ---- folded ffn + concat + output projection + relu ----
  gemm0(stream, aggCb, EE * HD, WeffB, HD, EE * HD, HD,
        out, nullptr, nullptr, 0, HD, NS, beff, nullptr, 0, 1);

  (void)in_sizes; (void)n_in; (void)out_size; (void)ws_size;
}
